// MultiHeadSelfAttention_83769042141193
// MI455X (gfx1250) — compile-verified, matrix operand order checked by reading, not run on hardware
//
#include <hip/hip_runtime.h>

// ---------------- types & helpers ----------------
typedef __attribute__((ext_vector_type(16))) __bf16         v16bf;
typedef __attribute__((ext_vector_type(8)))  float          v8f;
typedef __attribute__((ext_vector_type(8)))  unsigned short v8us;

struct Frag32 { v8us lo, hi; };

__device__ __forceinline__ v16bf make_frag(v8us lo, v8us hi) {
    Frag32 f{lo, hi};
    return __builtin_bit_cast(v16bf, f);
}

__device__ __forceinline__ unsigned short f32_to_bf16(float f) {
    unsigned int u = __builtin_bit_cast(unsigned int, f);
    u += 0x7fffu + ((u >> 16) & 1u);       // round-to-nearest-even
    return (unsigned short)(u >> 16);
}

// gfx1250 async global->LDS copy (ASYNCcnt-tracked, no VGPR round trip) with
// a portable fallback (load + ds_store). Probe result: builtin exists, takes
// (v4i32 AS1*, v4i32 AS3*, imm offset, imm cpol).
#if defined(__AMDGCN__) && __has_builtin(__builtin_amdgcn_global_load_async_to_lds_b128)
#define ATTN_ASYNC_COPY 1
#else
#define ATTN_ASYNC_COPY 0
#endif

typedef int v4i32_t __attribute__((vector_size(16)));

__device__ __forceinline__ void copy16_g2s(const unsigned short* g, unsigned short* s) {
#if ATTN_ASYNC_COPY
    __builtin_amdgcn_global_load_async_to_lds_b128(
        (__attribute__((address_space(1))) v4i32_t*)(g),
        (__attribute__((address_space(3))) v4i32_t*)(s), 0, 0);
#else
    *(v8us*)s = *(const v8us*)g;
#endif
}

__device__ __forceinline__ void wait_async_all() {
#if ATTN_ASYNC_COPY
    asm volatile("s_wait_asynccnt 0" ::: "memory");
#endif
}

// fixed problem dims
constexpr int Dm    = 1024;   // d_model
constexpr int Hn    = 16;     // heads
constexpr int DK    = 64;     // head dim
constexpr int Tn    = 2048;   // seq len
constexpr int Bn    = 4;      // batch
constexpr int Mrows = Bn * Tn; // 8192

// ---------------- conversion kernels ----------------
__global__ __launch_bounds__(256) void cvt_bf16(const float* __restrict__ in,
                                                unsigned short* __restrict__ out, int n) {
    int i = blockIdx.x * 256 + threadIdx.x;
    if (i < n) out[i] = f32_to_bf16(in[i]);
}

// w is [K=1024][N=1024]; store transposed wt[n][k]
__global__ __launch_bounds__(256) void cvt_wT(const float* __restrict__ w,
                                              unsigned short* __restrict__ wt) {
    int i = blockIdx.x * 256 + threadIdx.x;   // 0 .. 1024*1024-1
    int k = i >> 10, n = i & 1023;
    wt[(size_t)n * 1024 + k] = f32_to_bf16(w[i]);
}

// ---------------- WMMA GEMM ----------------
// C[M,1024] = A[M,1024](bf16) x B (stored Bt[n][k] bf16) + bias[n]
// block tile 128x128, 8 waves (2x4), wave tile 64x32, BK = 32,
// double-buffered LDS, software-pipelined staging.
// MODE 0: bf16 head-major out[(b*H+h)*T + t][d]   (value * mul)
// MODE 1: bf16 head-major transposed out[(b*H+h)*DK + d][t]
// MODE 2: fp32 row-major out[m][n]
template <int MODE>
__global__ __launch_bounds__(256) void gemm_bf16_wmma(
    const unsigned short* __restrict__ A,
    const unsigned short* __restrict__ Bt,
    const float* __restrict__ bias,
    void* __restrict__ out, float mul)
{
    __shared__ unsigned short As[2][128][40];   // 40-half rows: 80 B, 16B-aligned
    __shared__ unsigned short Bs[2][128][40];

    const int tid   = threadIdx.x;
    const int lane  = tid & 31, wave = tid >> 5;
    const int waveM = wave >> 2;      // 0..1 -> 64 rows each
    const int waveN = wave & 3;       // 0..3 -> 32 cols each
    const int m0 = blockIdx.y * 128;
    const int n0 = blockIdx.x * 128;
    const int rA = lane & 15, hg = lane >> 4;

    auto stage = [&](int kb, int buf) {
        const int k0 = kb * 32;
        #pragma unroll
        for (int i = 0; i < 2; ++i) {
            int c = tid + i * 256;                  // 512 chunks of 8 halfs each
            copy16_g2s(&A [(size_t)(m0 + (c >> 2)) * Dm + k0 + (c & 3) * 8],
                       &As[buf][c >> 2][(c & 3) * 8]);
            copy16_g2s(&Bt[(size_t)(n0 + (c >> 2)) * Dm + k0 + (c & 3) * 8],
                       &Bs[buf][c >> 2][(c & 3) * 8]);
        }
    };

    v8f acc[4][2] = {};

    constexpr int NK = Dm / 32;       // 32 k-steps
    stage(0, 0);
    wait_async_all();
    __syncthreads();

    for (int kb = 0; kb < NK; ++kb) {
        const int buf = kb & 1;
        if (kb + 1 < NK) stage(kb + 1, buf ^ 1);
        if (kb + 2 < NK) {            // L2 prefetch two tiles ahead
            const int k2 = (kb + 2) * 32;
            __builtin_prefetch(&A [(size_t)(m0 + (tid >> 2)) * Dm + k2 + (tid & 3) * 8], 0, 3);
            __builtin_prefetch(&Bt[(size_t)(n0 + (tid >> 2)) * Dm + k2 + (tid & 3) * 8], 0, 3);
        }

        v16bf aF[4], bF[2];
        #pragma unroll
        for (int i = 0; i < 4; ++i) {
            int r = waveM * 64 + i * 16 + rA;
            aF[i] = make_frag(*(const v8us*)&As[buf][r][hg * 8],
                              *(const v8us*)&As[buf][r][16 + hg * 8]);
        }
        #pragma unroll
        for (int j = 0; j < 2; ++j) {
            int r = waveN * 32 + j * 16 + rA;
            bF[j] = make_frag(*(const v8us*)&Bs[buf][r][hg * 8],
                              *(const v8us*)&Bs[buf][r][16 + hg * 8]);
        }
        #pragma unroll
        for (int i = 0; i < 4; ++i)
            #pragma unroll
            for (int j = 0; j < 2; ++j)
                acc[i][j] = __builtin_amdgcn_wmma_f32_16x16x32_bf16(
                    false, aF[i], false, bF[j], (short)0, acc[i][j], false, false);

        wait_async_all();             // next-tile copies done before buffer swap
        __syncthreads();
    }

    // epilogue: lane holds (m = r + 8*hg, n = rA) of each 16x16 tile
    #pragma unroll
    for (int i = 0; i < 4; ++i) {
        #pragma unroll
        for (int j = 0; j < 2; ++j) {
            int gn = n0 + waveN * 32 + j * 16 + rA;
            float bv = bias[gn];
            #pragma unroll
            for (int r = 0; r < 8; ++r) {
                int gm = m0 + waveM * 64 + i * 16 + r + 8 * hg;
                float val = acc[i][j][r] + bv;
                if constexpr (MODE == 0) {
                    int b = gm / Tn, t = gm % Tn, hh = gn / DK, d = gn % DK;
                    ((unsigned short*)out)[((size_t)(b * Hn + hh) * Tn + t) * DK + d] =
                        f32_to_bf16(val * mul);
                } else if constexpr (MODE == 1) {
                    int b = gm / Tn, t = gm % Tn, hh = gn / DK, d = gn % DK;
                    ((unsigned short*)out)[((size_t)(b * Hn + hh) * DK + d) * Tn + t] =
                        f32_to_bf16(val);
                } else {
                    ((float*)out)[(size_t)gm * Dm + gn] = val;
                }
            }
        }
    }
}

// ---------------- causal flash attention ----------------
// grid (T/64, B*H), 128 threads (4 waves). Block owns 64 query rows; all waves
// run a uniform kv loop over [0, q0+64) in steps of 32 so K/V tiles can be
// double-buffered in LDS cooperatively; fully-masked tiles are skipped
// wave-uniformly (EXEC stays all-ones around WMMA).
// Q pre-scaled by 1/sqrt(dk). Q,K: [B*H][T][64] bf16; Vt: [B*H][64][T] bf16.
__global__ __launch_bounds__(128) void attn_fwd(
    const unsigned short* __restrict__ Q,
    const unsigned short* __restrict__ Km,
    const unsigned short* __restrict__ Vt,
    unsigned short* __restrict__ Y)
{
    __shared__ unsigned short Klds[2][32][72];  // 32 kv rows x 64 dk (+8 pad)
    __shared__ unsigned short Vlds[2][64][40];  // 64 dk rows x 32 t  (+8 pad)
    __shared__ unsigned short Plds[4][16][40];  // per-wave P staging (16x32)

    const int bh = blockIdx.y;
    const int b = bh / Hn, h = bh % Hn;
    const int q0 = blockIdx.x * 64;
    const int wave = threadIdx.x >> 5, lane = threadIdx.x & 31;
    const int qbase = q0 + wave * 16;
    const int rA = lane & 15, hg = lane >> 4;

    const unsigned short* Qb = Q  + (size_t)bh * Tn * DK;
    const unsigned short* Kb = Km + (size_t)bh * Tn * DK;
    const unsigned short* Vb = Vt + (size_t)bh * DK * Tn;

    auto stage = [&](int s) {
        const int j0 = s * 32, buf = s & 1;
        #pragma unroll
        for (int i = 0; i < 2; ++i) {
            int cK = threadIdx.x + i * 128;     // 256 chunks: 32 rows x 8
            copy16_g2s(&Kb[(size_t)(j0 + (cK >> 3)) * DK + (cK & 7) * 8],
                       &Klds[buf][cK >> 3][(cK & 7) * 8]);
            int cV = threadIdx.x + i * 128;     // 256 chunks: 64 rows x 4
            copy16_g2s(&Vb[(size_t)(cV >> 2) * Tn + j0 + (cV & 3) * 8],
                       &Vlds[buf][cV >> 2][(cV & 3) * 8]);
        }
    };

    // preload Q A-fragments for the two 32-wide chunks of dk=64
    v16bf qf[2];
    #pragma unroll
    for (int kk = 0; kk < 2; ++kk) {
        const unsigned short* p = &Qb[(size_t)(qbase + rA) * DK + kk * 32 + hg * 8];
        qf[kk] = make_frag(*(const v8us*)p, *(const v8us*)(p + 16));
    }

    v8f o[4] = {};
    float m_i[8], l_i[8];
    #pragma unroll
    for (int r = 0; r < 8; ++r) { m_i[r] = -1e30f; l_i[r] = 0.f; }

    const int steps = (q0 + 64) / 32;           // uniform across block
    stage(0);
    wait_async_all();
    __syncthreads();

    for (int s = 0; s < steps; ++s) {
        const int j0 = s * 32, buf = s & 1;
        if (s + 1 < steps) stage(s + 1);
        if (s + 2 < steps) {                    // L2 prefetch two tiles ahead
            const int j2 = j0 + 64;
            __builtin_prefetch(&Kb[(size_t)(j2 + (threadIdx.x >> 3)) * DK + (threadIdx.x & 7) * 8], 0, 3);
            __builtin_prefetch(&Vb[(size_t)(threadIdx.x >> 2) * Tn + j2 + (threadIdx.x & 3) * 8], 0, 3);
        }

        if (j0 <= qbase + 15) {                 // wave-uniform skip of masked tiles
            // S = Q K^T for 16 rows x 32 cols (two 16-col tiles)
            v8f sc[2];
            #pragma unroll
            for (int c = 0; c < 2; ++c) {
                v8f accs = {};
                #pragma unroll
                for (int kk = 0; kk < 2; ++kk) {
                    int kr = c * 16 + rA;
                    v16bf kf = make_frag(*(const v8us*)&Klds[buf][kr][kk * 32 + hg * 8],
                                         *(const v8us*)&Klds[buf][kr][kk * 32 + 16 + hg * 8]);
                    accs = __builtin_amdgcn_wmma_f32_16x16x32_bf16(
                        false, qf[kk], false, kf, (short)0, accs, false, false);
                }
                sc[c] = accs;
            }

            // causal mask: (row = qbase + r + 8*hg, col = j0 + c*16 + rA)
            if (j0 + 31 > qbase) {
                #pragma unroll
                for (int c = 0; c < 2; ++c)
                    #pragma unroll
                    for (int r = 0; r < 8; ++r)
                        if (j0 + c * 16 + rA > qbase + r + 8 * hg) sc[c][r] = -1e30f;
            }

            // online softmax (row stats across 16-lane groups)
            float corr[8];
            #pragma unroll
            for (int r = 0; r < 8; ++r) {
                float mx = fmaxf(sc[0][r], sc[1][r]);
                #pragma unroll
                for (int off = 1; off < 16; off <<= 1)
                    mx = fmaxf(mx, __shfl_xor(mx, off, 32));
                float mnew = fmaxf(m_i[r], mx);
                corr[r] = __expf(m_i[r] - mnew);
                float p0 = __expf(sc[0][r] - mnew);
                float p1 = __expf(sc[1][r] - mnew);
                sc[0][r] = p0; sc[1][r] = p1;
                float rs = p0 + p1;
                #pragma unroll
                for (int off = 1; off < 16; off <<= 1)
                    rs += __shfl_xor(rs, off, 32);
                l_i[r] = l_i[r] * corr[r] + rs;
                m_i[r] = mnew;
            }
            #pragma unroll
            for (int c4 = 0; c4 < 4; ++c4)
                #pragma unroll
                for (int r = 0; r < 8; ++r)
                    o[c4][r] *= corr[r];

            // redistribute P (C-layout -> A-fragment layout) via per-wave LDS
            #pragma unroll
            for (int c = 0; c < 2; ++c)
                #pragma unroll
                for (int r = 0; r < 8; ++r)
                    Plds[wave][r + 8 * hg][c * 16 + rA] = f32_to_bf16(sc[c][r]);
            asm volatile("s_wait_dscnt 0" ::: "memory");  // DS in-order per wave
            v16bf pf = make_frag(*(const v8us*)&Plds[wave][rA][hg * 8],
                                 *(const v8us*)&Plds[wave][rA][16 + hg * 8]);

            // O += P(16x32) x V(32x64)
            #pragma unroll
            for (int c4 = 0; c4 < 4; ++c4) {
                int vr = c4 * 16 + rA;
                v16bf vf = make_frag(*(const v8us*)&Vlds[buf][vr][hg * 8],
                                     *(const v8us*)&Vlds[buf][vr][16 + hg * 8]);
                o[c4] = __builtin_amdgcn_wmma_f32_16x16x32_bf16(
                    false, pf, false, vf, (short)0, o[c4], false, false);
            }
        }

        wait_async_all();
        __syncthreads();
    }

    // normalize and write Y (bf16, [B*T, 1024])
    #pragma unroll
    for (int c4 = 0; c4 < 4; ++c4) {
        #pragma unroll
        for (int r = 0; r < 8; ++r) {
            float inv = 1.0f / l_i[r];
            int row = qbase + r + 8 * hg;
            int col = h * DK + c4 * 16 + rA;
            Y[((size_t)(b * Tn) + row) * Dm + col] = f32_to_bf16(o[c4][r] * inv);
        }
    }
}

// ---------------- host launch ----------------
extern "C" void kernel_launch(void* const* d_in, const int* in_sizes, int n_in,
                              void* d_out, int out_size, void* d_ws, size_t ws_size,
                              hipStream_t stream) {
    const float* x  = (const float*)d_in[0];
    const float* wq = (const float*)d_in[1];
    const float* bq = (const float*)d_in[2];
    const float* wk = (const float*)d_in[3];
    const float* bk = (const float*)d_in[4];
    const float* wv = (const float*)d_in[5];
    const float* bv = (const float*)d_in[6];
    const float* wo = (const float*)d_in[7];
    const float* bo = (const float*)d_in[8];

    char* ws = (char*)d_ws;
    unsigned short* xb  = (unsigned short*)(ws);                    // 16 MB
    unsigned short* wqT = (unsigned short*)(ws + 16777216);         //  2 MB
    unsigned short* wkT = (unsigned short*)(ws + 18874368);
    unsigned short* wvT = (unsigned short*)(ws + 20971520);
    unsigned short* woT = (unsigned short*)(ws + 23068672);
    unsigned short* qb  = (unsigned short*)(ws + 25165824);         // 16 MB
    unsigned short* kb  = (unsigned short*)(ws + 41943040);         // 16 MB
    unsigned short* vt  = (unsigned short*)(ws + 58720256);         // 16 MB
    unsigned short* yb  = (unsigned short*)(ws + 75497472);         // 16 MB

    const int nX = Mrows * Dm;          // 8192*1024
    cvt_bf16<<<nX / 256, 256, 0, stream>>>(x, xb, nX);
    cvt_wT<<<(Dm * Dm) / 256, 256, 0, stream>>>(wq, wqT);
    cvt_wT<<<(Dm * Dm) / 256, 256, 0, stream>>>(wk, wkT);
    cvt_wT<<<(Dm * Dm) / 256, 256, 0, stream>>>(wv, wvT);
    cvt_wT<<<(Dm * Dm) / 256, 256, 0, stream>>>(wo, woT);

    dim3 gg(Dm / 128, Mrows / 128);     // (8, 64)
    gemm_bf16_wmma<0><<<gg, 256, 0, stream>>>(xb, wqT, bq, qb, 0.125f); // Q * 1/sqrt(64)
    gemm_bf16_wmma<0><<<gg, 256, 0, stream>>>(xb, wkT, bk, kb, 1.0f);  // K
    gemm_bf16_wmma<1><<<gg, 256, 0, stream>>>(xb, wvT, bv, vt, 1.0f);  // V (transposed)

    dim3 ga(Tn / 64, Bn * Hn);          // (32, 64)
    attn_fwd<<<ga, 128, 0, stream>>>(qb, kb, vt, yb);

    gemm_bf16_wmma<2><<<gg, 256, 0, stream>>>(yb, woT, bo, d_out, 1.0f); // final, fp32
}
